// part_attention_22969485099046
// MI455X (gfx1250) — compile-verified, matrix operand order checked by reading, not run on hardware
//
#include <hip/hip_runtime.h>

// CDNA5 / gfx1250 — wave32, WMMA 16x16x32 bf16
typedef __attribute__((ext_vector_type(16))) __bf16 v16bf;
typedef __attribute__((ext_vector_type(8)))  float  v8f;

#define C_DIM   768            // input channels
#define D_DIM   64             // projection dim
#define TILE_B  16             // batch rows per workgroup
#define KCHUNKS (C_DIM / 32)   // 24 k-steps of 32
// bf16 weight-fragment buffer: [4 proj][24 kchunk][4 ntile][32 lane][16 elem]
#define WFRAG_ELEMS (4 * KCHUNKS * 4 * 32 * 16)
#define WFRAG_BYTES (WFRAG_ELEMS * 2)

// ---------------------------------------------------------------------------
// Pre-pass: convert fp32 weights -> bf16, swizzled into per-lane WMMA
// B-fragment layout so the main loop does one 32B contiguous load per frag.
// gid = ((proj*24 + kc)*4 + t)*32 + lane ; each thread emits 16 bf16.
// K mapping inside a fragment (wave32 16-bit B layout):
//   i<8  -> k = kc*32 + 8*half + i
//   i>=8 -> k = kc*32 + 16 + 8*half + (i-8)
// ---------------------------------------------------------------------------
__global__ __launch_bounds__(256) void precvt_weights(
    const float* __restrict__ w_whole, const float* __restrict__ w_head,
    const float* __restrict__ w_upper, const float* __restrict__ w_lower,
    __bf16* __restrict__ wfrag)
{
    int gid  = blockIdx.x * 256 + threadIdx.x;      // 0 .. 12287
    int lane = gid & 31;
    int rest = gid >> 5;                            // 0 .. 383
    int t    = rest & 3;
    rest >>= 2;                                     // 0 .. 95
    int kc   = rest % KCHUNKS;
    int proj = rest / KCHUNKS;

    const float* w = (proj == 0) ? w_whole : (proj == 1) ? w_head
                    : (proj == 2) ? w_upper : w_lower;

    int lm = lane & 15, half = lane >> 4;
    const float* src = w + (size_t)(16 * t + lm) * C_DIM + kc * 32 + 8 * half;
    __bf16* dst = wfrag + (size_t)gid * 16;
    #pragma unroll
    for (int i = 0; i < 8; ++i) dst[i]     = (__bf16)src[i];
    #pragma unroll
    for (int i = 0; i < 8; ++i) dst[8 + i] = (__bf16)src[16 + i];
}

// ---------------------------------------------------------------------------
// Main kernel (fast path): 4 wave32s per block, one projection per wave.
// B fragments loaded pre-swizzled in bf16 from d_ws (L2-resident).
// ---------------------------------------------------------------------------
__global__ __launch_bounds__(128) void part_attn_frag(
    const float* __restrict__ x_whole, const float* __restrict__ x_head,
    const float* __restrict__ x_upper, const float* __restrict__ x_lower,
    const __bf16* __restrict__ wfrag,
    const float* __restrict__ b_whole, const float* __restrict__ b_head,
    const float* __restrict__ b_upper, const float* __restrict__ b_lower,
    float* __restrict__ out)
{
    __shared__ float vmat[4][TILE_B][D_DIM];   // 16 KB

    const int tid  = threadIdx.x;
    const int wave = tid >> 5;          // projection id
    const int lane = tid & 31;
    const int half = lane >> 4;
    const int lm   = lane & 15;
    const int row0 = blockIdx.x * TILE_B;

    const float* xbase = (wave == 0) ? x_whole : (wave == 1) ? x_head
                        : (wave == 2) ? x_upper : x_lower;
    const float* bbase = (wave == 0) ? b_whole : (wave == 1) ? b_head
                        : (wave == 2) ? b_upper : b_lower;

    const float*  xp  = xbase + (size_t)(row0 + lm) * C_DIM + 8 * half;
    const __bf16* wfp = wfrag + (size_t)wave * KCHUNKS * 4 * 32 * 16
                              + (size_t)lane * 16;

    v8f acc[4] = {};

    for (int kc = 0; kc < KCHUNKS; ++kc) {
        __builtin_prefetch(xp + kc * 32 + 32, 0, 1);

        // A fragment: two aligned 32B fp32 chunks -> bf16
        const float4* pa = (const float4*)(xp + kc * 32);
        float4 a0 = pa[0], a1 = pa[1];      // K = 8*half + 0..7
        float4 a2 = pa[4], a3 = pa[5];      // K = 16 + 8*half + 0..7
        v16bf A;
        A[0]=(__bf16)a0.x;  A[1]=(__bf16)a0.y;  A[2]=(__bf16)a0.z;  A[3]=(__bf16)a0.w;
        A[4]=(__bf16)a1.x;  A[5]=(__bf16)a1.y;  A[6]=(__bf16)a1.z;  A[7]=(__bf16)a1.w;
        A[8]=(__bf16)a2.x;  A[9]=(__bf16)a2.y;  A[10]=(__bf16)a2.z; A[11]=(__bf16)a2.w;
        A[12]=(__bf16)a3.x; A[13]=(__bf16)a3.y; A[14]=(__bf16)a3.z; A[15]=(__bf16)a3.w;

        // B fragments: single contiguous 32B bf16 load per n-tile, no converts
        #pragma unroll
        for (int t = 0; t < 4; ++t) {
            v16bf Bf = *(const v16bf*)(wfp + (size_t)(kc * 4 + t) * (32 * 16));
            acc[t] = __builtin_amdgcn_wmma_f32_16x16x32_bf16(
                false, A, false, Bf, (short)0, acc[t], false, false);
        }
    }

    // Bias add + stage v_p tile to LDS (C/D layout: VGPR r -> M=r+8*half, N=lm)
    #pragma unroll
    for (int t = 0; t < 4; ++t) {
        float bv = bbase[16 * t + lm];
        #pragma unroll
        for (int r = 0; r < 8; ++r)
            vmat[wave][r + 8 * half][16 * t + lm] = acc[t][r] + bv;
    }
    __syncthreads();

    if (tid < TILE_B) {
        const int m = tid;
        float dh = 0.f, du = 0.f, dl = 0.f;
        #pragma unroll 8
        for (int d = 0; d < D_DIM; ++d) {
            float vw = vmat[0][m][d];
            dh = fmaf(vmat[1][m][d], vw, dh);
            du = fmaf(vmat[2][m][d], vw, du);
            dl = fmaf(vmat[3][m][d], vw, dl);
        }
        float mx = fmaxf(dh, fmaxf(du, dl));
        float e0 = __expf(dh - mx);
        float e1 = __expf(du - mx);
        float e2 = __expf(dl - mx);
        float inv = 1.0f / (e0 + e1 + e2);
        float* o = out + (size_t)(row0 + m) * 3;
        o[0] = e0 * inv;
        o[1] = e1 * inv;
        o[2] = e2 * inv;
    }
}

// ---------------------------------------------------------------------------
// Fallback (ws too small): convert weights in-loop (round-1 kernel).
// ---------------------------------------------------------------------------
__global__ __launch_bounds__(128) void part_attn_cvt(
    const float* __restrict__ x_whole, const float* __restrict__ x_head,
    const float* __restrict__ x_upper, const float* __restrict__ x_lower,
    const float* __restrict__ w_whole, const float* __restrict__ b_whole,
    const float* __restrict__ w_head,  const float* __restrict__ b_head,
    const float* __restrict__ w_upper, const float* __restrict__ b_upper,
    const float* __restrict__ w_lower, const float* __restrict__ b_lower,
    float* __restrict__ out)
{
    __shared__ float vmat[4][TILE_B][D_DIM];

    const int tid  = threadIdx.x;
    const int wave = tid >> 5;
    const int lane = tid & 31;
    const int half = lane >> 4;
    const int lm   = lane & 15;
    const int row0 = blockIdx.x * TILE_B;

    const float* xbase = (wave == 0) ? x_whole : (wave == 1) ? x_head
                        : (wave == 2) ? x_upper : x_lower;
    const float* wbase = (wave == 0) ? w_whole : (wave == 1) ? w_head
                        : (wave == 2) ? w_upper : w_lower;
    const float* bbase = (wave == 0) ? b_whole : (wave == 1) ? b_head
                        : (wave == 2) ? b_upper : b_lower;

    const float* xp = xbase + (size_t)(row0 + lm) * C_DIM + 8 * half;
    v8f acc[4] = {};

    for (int k0 = 0; k0 < C_DIM; k0 += 32) {
        const float4* pa = (const float4*)(xp + k0);
        float4 a0 = pa[0], a1 = pa[1], a2 = pa[4], a3 = pa[5];
        v16bf A;
        A[0]=(__bf16)a0.x;  A[1]=(__bf16)a0.y;  A[2]=(__bf16)a0.z;  A[3]=(__bf16)a0.w;
        A[4]=(__bf16)a1.x;  A[5]=(__bf16)a1.y;  A[6]=(__bf16)a1.z;  A[7]=(__bf16)a1.w;
        A[8]=(__bf16)a2.x;  A[9]=(__bf16)a2.y;  A[10]=(__bf16)a2.z; A[11]=(__bf16)a2.w;
        A[12]=(__bf16)a3.x; A[13]=(__bf16)a3.y; A[14]=(__bf16)a3.z; A[15]=(__bf16)a3.w;

        #pragma unroll
        for (int t = 0; t < 4; ++t) {
            const float4* pb = (const float4*)(wbase + (size_t)(16 * t + lm) * C_DIM
                                               + k0 + 8 * half);
            float4 b0 = pb[0], b1 = pb[1], b2 = pb[4], b3 = pb[5];
            v16bf Bf;
            Bf[0]=(__bf16)b0.x;  Bf[1]=(__bf16)b0.y;  Bf[2]=(__bf16)b0.z;  Bf[3]=(__bf16)b0.w;
            Bf[4]=(__bf16)b1.x;  Bf[5]=(__bf16)b1.y;  Bf[6]=(__bf16)b1.z;  Bf[7]=(__bf16)b1.w;
            Bf[8]=(__bf16)b2.x;  Bf[9]=(__bf16)b2.y;  Bf[10]=(__bf16)b2.z; Bf[11]=(__bf16)b2.w;
            Bf[12]=(__bf16)b3.x; Bf[13]=(__bf16)b3.y; Bf[14]=(__bf16)b3.z; Bf[15]=(__bf16)b3.w;
            acc[t] = __builtin_amdgcn_wmma_f32_16x16x32_bf16(
                false, A, false, Bf, (short)0, acc[t], false, false);
        }
    }

    #pragma unroll
    for (int t = 0; t < 4; ++t) {
        float bv = bbase[16 * t + lm];
        #pragma unroll
        for (int r = 0; r < 8; ++r)
            vmat[wave][r + 8 * half][16 * t + lm] = acc[t][r] + bv;
    }
    __syncthreads();

    if (tid < TILE_B) {
        const int m = tid;
        float dh = 0.f, du = 0.f, dl = 0.f;
        #pragma unroll 8
        for (int d = 0; d < D_DIM; ++d) {
            float vw = vmat[0][m][d];
            dh = fmaf(vmat[1][m][d], vw, dh);
            du = fmaf(vmat[2][m][d], vw, du);
            dl = fmaf(vmat[3][m][d], vw, dl);
        }
        float mx = fmaxf(dh, fmaxf(du, dl));
        float e0 = __expf(dh - mx);
        float e1 = __expf(du - mx);
        float e2 = __expf(dl - mx);
        float inv = 1.0f / (e0 + e1 + e2);
        float* o = out + (size_t)(row0 + m) * 3;
        o[0] = e0 * inv;
        o[1] = e1 * inv;
        o[2] = e2 * inv;
    }
}

extern "C" void kernel_launch(void* const* d_in, const int* in_sizes, int n_in,
                              void* d_out, int out_size, void* d_ws, size_t ws_size,
                              hipStream_t stream) {
    (void)n_in; (void)out_size;
    const float* x_whole = (const float*)d_in[0];
    const float* x_head  = (const float*)d_in[1];
    const float* x_upper = (const float*)d_in[2];
    const float* x_lower = (const float*)d_in[3];
    const float* w_whole = (const float*)d_in[4];
    const float* b_whole = (const float*)d_in[5];
    const float* w_head  = (const float*)d_in[6];
    const float* b_head  = (const float*)d_in[7];
    const float* w_upper = (const float*)d_in[8];
    const float* b_upper = (const float*)d_in[9];
    const float* w_lower = (const float*)d_in[10];
    const float* b_lower = (const float*)d_in[11];
    float* out = (float*)d_out;

    const int Bsz  = in_sizes[0] / C_DIM;   // 8192
    const int grid = Bsz / TILE_B;          // 512

    if (ws_size >= (size_t)WFRAG_BYTES) {
        __bf16* wfrag = (__bf16*)d_ws;
        // 12288 threads: swizzle+convert all 4 weight matrices to bf16 fragments
        precvt_weights<<<48, 256, 0, stream>>>(w_whole, w_head, w_upper, w_lower, wfrag);
        part_attn_frag<<<grid, 128, 0, stream>>>(
            x_whole, x_head, x_upper, x_lower, wfrag,
            b_whole, b_head, b_upper, b_lower, out);
    } else {
        part_attn_cvt<<<grid, 128, 0, stream>>>(
            x_whole, x_head, x_upper, x_lower,
            w_whole, b_whole, w_head, b_head,
            w_upper, b_upper, w_lower, b_lower, out);
    }
}
